// AttentionModel_39848706573344
// MI455X (gfx1250) — compile-verified, matrix operand order checked by reading, not run on hardware
//
#include <hip/hip_runtime.h>

// ---------------------------------------------------------------------------
// Problem constants (from reference): B=2, S=2048, E=1024, H=16, D=64
// ---------------------------------------------------------------------------
#define BB 2
#define SS 2048
#define EE 1024
#define HH 16
#define DD 64

typedef _Float16 f16;
typedef __attribute__((ext_vector_type(16))) _Float16 v16h;
typedef __attribute__((ext_vector_type(8)))  _Float16 v8h;
typedef __attribute__((ext_vector_type(8)))  float    v8f;
typedef __attribute__((ext_vector_type(4)))  int      v4i;

// Pointer types matching the async-to-LDS builtin signature:
//   (int4 addrspace(1)*, int4 addrspace(3)*, imm offset, imm cpol)
typedef __attribute__((address_space(1))) v4i* gv4i_p;
typedef __attribute__((address_space(3))) v4i* lv4i_p;

// Optional CDNA5 async global->LDS copy path (probe via __has_builtin so the
// build never breaks if the toolchain lacks it).
#if defined(__has_builtin)
#if __has_builtin(__builtin_amdgcn_global_load_async_to_lds_b128) && \
    __has_builtin(__builtin_amdgcn_s_wait_asynccnt)
#define HAVE_ASYNC_LDS 1
#endif
#endif

// D = A(16x32 f16) * B(32x16 f16) + C(16x16 f32)
__device__ __forceinline__ v8f wmma_f16(v16h a, v16h b, v8f c) {
  return __builtin_amdgcn_wmma_f32_16x16x32_f16(
      /*neg_a=*/false, a, /*neg_b=*/false, b,
      /*c_mod=*/(short)0, c, /*reuse_a=*/false, /*reuse_b=*/false);
}

// Per-lane fragment gather for a row-major 16-bit operand (CDNA5 16x32 A
// layout; B 32x16 mirrors it with N in place of M):
//   lane half h holds K = kbase + h*8 .. +7   and   kbase + 16 + h*8 .. +7
// i.e. two contiguous 16-byte chunks per lane.
__device__ __forceinline__ v16h load_frag(const f16* row, int kbase, int half) {
  v8h lo = *(const v8h*)(row + kbase + half * 8);
  v8h hi = *(const v8h*)(row + kbase + 16 + half * 8);
  return __builtin_shufflevector(lo, hi, 0, 1, 2, 3, 4, 5, 6, 7,
                                 8, 9, 10, 11, 12, 13, 14, 15);
}

// ---------------------------------------------------------------------------
// Kernel 1: y = x @ W^T + b  for W in {Wq, Wk, Wv}  (blockIdx.y selects)
//   Each wave computes a 32x64 strip (8 accumulators).  K-loop step 32 with
//   register double-buffering: next step's fragments are loaded before the
//   current step's 8 WMMAs so loads overlap matrix math.
// ---------------------------------------------------------------------------
__global__ __launch_bounds__(128) void qkv_gemm(
    const f16* __restrict__ x,
    const f16* __restrict__ Wq, const f16* __restrict__ bq,
    const f16* __restrict__ Wk, const f16* __restrict__ bk,
    const f16* __restrict__ Wv, const f16* __restrict__ bv,
    f16* __restrict__ ws) {
  const int mat = blockIdx.y;
  const f16* W    = (mat == 0) ? Wq : (mat == 1) ? Wk : Wv;
  const f16* bias = (mat == 0) ? bq : (mat == 1) ? bk : bv;
  f16* y = ws + (size_t)mat * ((size_t)BB * SS * EE);

  const int tid    = threadIdx.x;
  const int lane   = tid & 31;
  const int lane16 = lane & 15;
  const int half   = lane >> 4;
  const int wv     = tid >> 5;

  const int w  = blockIdx.x * 4 + wv;     // flat wave-tile id, [0, 2048)
  const int mt = w >> 4;                  // 32-row tile,   [0, 128)
  const int nt = w & 15;                  // 64-col tile,   [0, 16)

  const f16* arow0 = x + (size_t)(mt * 32 + lane16) * EE;
  const f16* arow1 = arow0 + (size_t)16 * EE;
  const f16* brow0 = W + (size_t)(nt * 64 +  0 + lane16) * EE;
  const f16* brow1 = W + (size_t)(nt * 64 + 16 + lane16) * EE;
  const f16* brow2 = W + (size_t)(nt * 64 + 32 + lane16) * EE;
  const f16* brow3 = W + (size_t)(nt * 64 + 48 + lane16) * EE;

  v8f acc[2][4];
#pragma unroll
  for (int i = 0; i < 2; ++i)
#pragma unroll
    for (int j = 0; j < 4; ++j) acc[i][j] = (v8f){};

  // prologue fragments
  v16h a0 = load_frag(arow0, 0, half);
  v16h a1 = load_frag(arow1, 0, half);
  v16h b0 = load_frag(brow0, 0, half);
  v16h b1 = load_frag(brow1, 0, half);
  v16h b2 = load_frag(brow2, 0, half);
  v16h b3 = load_frag(brow3, 0, half);

  for (int kk = 0; kk < EE; kk += 32) {
    // prefetch next step (wraps to 0 on the final step; result unused)
    const int kn = (kk + 32 < EE) ? (kk + 32) : 0;
    v16h na0 = load_frag(arow0, kn, half);
    v16h na1 = load_frag(arow1, kn, half);
    v16h nb0 = load_frag(brow0, kn, half);
    v16h nb1 = load_frag(brow1, kn, half);
    v16h nb2 = load_frag(brow2, kn, half);
    v16h nb3 = load_frag(brow3, kn, half);

    acc[0][0] = wmma_f16(a0, b0, acc[0][0]);
    acc[1][0] = wmma_f16(a1, b0, acc[1][0]);
    acc[0][1] = wmma_f16(a0, b1, acc[0][1]);
    acc[1][1] = wmma_f16(a1, b1, acc[1][1]);
    acc[0][2] = wmma_f16(a0, b2, acc[0][2]);
    acc[1][2] = wmma_f16(a1, b2, acc[1][2]);
    acc[0][3] = wmma_f16(a0, b3, acc[0][3]);
    acc[1][3] = wmma_f16(a1, b3, acc[1][3]);

    a0 = na0; a1 = na1;
    b0 = nb0; b1 = nb1; b2 = nb2; b3 = nb3;
  }

  // bias add (fp32) + fp16 store. Accumulator map: VGPR r, lane L ->
  // m = r + 8*(L/16), n = L%16 (per sub-tile).
#pragma unroll
  for (int mi = 0; mi < 2; ++mi) {
#pragma unroll
    for (int s = 0; s < 4; ++s) {
      const int n = nt * 64 + s * 16 + lane16;
      const float bf = (float)bias[n];
#pragma unroll
      for (int r = 0; r < 8; ++r) {
        const int m = mt * 32 + mi * 16 + r + 8 * half;
        y[(size_t)m * EE + n] = (f16)(acc[mi][s][r] + bf);
      }
    }
  }
}

// ---------------------------------------------------------------------------
// Kernel 2: flash attention per (b, h, 64-query block).
//   128 threads = 4 waves; each wave owns 16 query rows.
//   64-key blocks staged through LDS (K async-to-LDS when available,
//   V transposed via register scatter).  16 WMMAs per barrier pair.
// ---------------------------------------------------------------------------
__global__ __launch_bounds__(128) void flash_attn(
    const f16* __restrict__ q, const f16* __restrict__ k,
    const f16* __restrict__ v, f16* __restrict__ out) {
  __shared__ f16 Ks[64 * 72];        // K block, row-major rows of 64 (+pad)
  __shared__ f16 VTs[64 * 72];       // V block transposed: VT[d][t] (+pad)
  __shared__ f16 Ps[4 * 16 * 72];    // per-wave probability tiles 16x64 (+pad)

  const int bid = blockIdx.x;
  const int qb  = bid & 31;          // query block   [0, 32)
  const int h   = (bid >> 5) & 15;   // head          [0, 16)
  const int b   = bid >> 9;          // batch         [0, 2)

  const int tid    = threadIdx.x;
  const int lane   = tid & 31;
  const int lane16 = lane & 15;
  const int half   = lane >> 4;
  const int wv     = tid >> 5;

  // --- Q fragments (resident in VGPRs), pre-scaled by 1/sqrt(D) = 0.125 ---
  const int s_q = qb * 64 + wv * 16 + lane16;
  const f16* qrow = q + (size_t)(b * SS + s_q) * EE + h * DD;
  v16h aq0 = load_frag(qrow, 0, half);
  v16h aq1 = load_frag(qrow, 32, half);
#pragma unroll
  for (int i = 0; i < 16; ++i) {
    aq0[i] = aq0[i] * (f16)0.125f;
    aq1[i] = aq1[i] * (f16)0.125f;
  }

  float m_s[8], l_s[8];
  v8f o[4];
#pragma unroll
  for (int dt = 0; dt < 4; ++dt) o[dt] = (v8f){};
#pragma unroll
  for (int r = 0; r < 8; ++r) { m_s[r] = -3.0e38f; l_s[r] = 0.0f; }

  f16* pw = &Ps[wv * 16 * 72];
  const size_t kvbase = (size_t)(b * SS) * EE + h * DD;

  for (int kb = 0; kb < SS / 64; ++kb) {
    __syncthreads();  // previous iteration's LDS reads complete

    // --- cooperative stage: 64x64 K rows + V transposed (512 chunks each) ---
#pragma unroll
    for (int c = tid; c < 512; c += 128) {
      const int row = c >> 3;             // key within block [0,64)
      const int off = (c & 7) * 8;        // dim offset       [0,64)
      const size_t g = kvbase + (size_t)(kb * 64 + row) * EE + off;
#if defined(HAVE_ASYNC_LDS)
      __builtin_amdgcn_global_load_async_to_lds_b128(
          (gv4i_p)(void*)(k + g),
          (lv4i_p)(void*)&Ks[row * 72 + off],
          /*offset=*/0, /*cpol=*/0);
#else
      *(v8h*)&Ks[row * 72 + off] = *(const v8h*)(k + g);
#endif
      const v8h vv = *(const v8h*)(v + g);
#pragma unroll
      for (int j = 0; j < 8; ++j) VTs[(off + j) * 72 + row] = vv[j];
      if (kb + 1 < SS / 64)  // hint next block into cache
        __builtin_prefetch(k + g + (size_t)64 * EE, 0, 1);
    }
#if defined(HAVE_ASYNC_LDS)
    __builtin_amdgcn_s_wait_asynccnt(0);
#endif
    __syncthreads();

    // --- S = Q * K^T : four 16x16 accumulators covering 64 keys ---
    v8f s[4];
#pragma unroll
    for (int nt = 0; nt < 4; ++nt) {
      const f16* krow = &Ks[(nt * 16 + lane16) * 72];
      s[nt] = (v8f){};
      s[nt] = wmma_f16(aq0, load_frag(krow, 0, half), s[nt]);
      s[nt] = wmma_f16(aq1, load_frag(krow, 32, half), s[nt]);
    }

    // --- online softmax (row reductions across 16-lane groups) ---
#pragma unroll
    for (int r = 0; r < 8; ++r) {
      const float v0 = s[0][r], v1 = s[1][r], v2 = s[2][r], v3 = s[3][r];
      float rm = fmaxf(fmaxf(v0, v1), fmaxf(v2, v3));
#pragma unroll
      for (int off = 1; off < 16; off <<= 1)
        rm = fmaxf(rm, __shfl_xor(rm, off, 32));
      const float mnew = fmaxf(m_s[r], rm);
      const float corr = __expf(m_s[r] - mnew);
      const float p0 = __expf(v0 - mnew);
      const float p1 = __expf(v1 - mnew);
      const float p2 = __expf(v2 - mnew);
      const float p3 = __expf(v3 - mnew);
      float rs = (p0 + p1) + (p2 + p3);
#pragma unroll
      for (int off = 1; off < 16; off <<= 1)
        rs += __shfl_xor(rs, off, 32);
      l_s[r] = l_s[r] * corr + rs;
      m_s[r] = mnew;
#pragma unroll
      for (int dt = 0; dt < 4; ++dt) o[dt][r] *= corr;
      const int m = r + 8 * half;
      pw[m * 72 +  0 + lane16] = (f16)p0;
      pw[m * 72 + 16 + lane16] = (f16)p1;
      pw[m * 72 + 32 + lane16] = (f16)p2;
      pw[m * 72 + 48 + lane16] = (f16)p3;
    }
    __syncthreads();  // P visible; Ks/VTs reads done before next stage

    // --- O += P * V : A = P tile (16x64), B = VT rows (contiguous) ---
    const v16h ap0 = load_frag(&pw[lane16 * 72], 0, half);
    const v16h ap1 = load_frag(&pw[lane16 * 72], 32, half);
#pragma unroll
    for (int dt = 0; dt < 4; ++dt) {
      const f16* vrow = &VTs[(dt * 16 + lane16) * 72];
      o[dt] = wmma_f16(ap0, load_frag(vrow, 0, half), o[dt]);
      o[dt] = wmma_f16(ap1, load_frag(vrow, 32, half), o[dt]);
    }
  }

  // --- epilogue: O / l, store fp16 to [B, S, E] ---
  float rinv[8];
#pragma unroll
  for (int r = 0; r < 8; ++r) rinv[r] = 1.0f / l_s[r];

#pragma unroll
  for (int dt = 0; dt < 4; ++dt) {
    const int e = h * DD + dt * 16 + lane16;
#pragma unroll
    for (int r = 0; r < 8; ++r) {
      const int s = qb * 64 + wv * 16 + r + 8 * half;
      out[(size_t)(b * SS + s) * EE + e] = (f16)(o[dt][r] * rinv[r]);
    }
  }
}

// ---------------------------------------------------------------------------
// Launch: inputs in setup_inputs() order: x, Wq, bq, Wk, bk, Wv, bv (all f16)
// Workspace: Q | K | V, each [B*S, E] f16 = 8 MiB (24 MiB total).
// ---------------------------------------------------------------------------
extern "C" void kernel_launch(void* const* d_in, const int* in_sizes, int n_in,
                              void* d_out, int out_size, void* d_ws,
                              size_t ws_size, hipStream_t stream) {
  (void)in_sizes; (void)n_in; (void)out_size; (void)ws_size;
  const f16* x  = (const f16*)d_in[0];
  const f16* Wq = (const f16*)d_in[1];
  const f16* bq = (const f16*)d_in[2];
  const f16* Wk = (const f16*)d_in[3];
  const f16* bk = (const f16*)d_in[4];
  const f16* Wv = (const f16*)d_in[5];
  const f16* bv = (const f16*)d_in[6];
  f16* out = (f16*)d_out;
  f16* ws  = (f16*)d_ws;

  f16* qws = ws;
  f16* kws = ws + (size_t)BB * SS * EE;
  f16* vws = ws + (size_t)2 * BB * SS * EE;

  // 3 projections: 2048 wave-tiles each (128 M-tiles x 16 N64-tiles), 4 waves/WG
  qkv_gemm<<<dim3(512, 3), 128, 0, stream>>>(x, Wq, bq, Wk, bk, Wv, bv, ws);

  // 32 query blocks x 16 heads x 2 batches = 1024 workgroups
  flash_attn<<<dim3(1024), 128, 0, stream>>>(qws, kws, vws, out);
}
